// MultiHeadAttention_979252543908
// MI455X (gfx1250) — compile-verified, matrix operand order checked by reading, not run on hardware
//
#include <hip/hip_runtime.h>

typedef __attribute__((ext_vector_type(16))) _Float16 v16h;
typedef __attribute__((ext_vector_type(8)))  _Float16 h8;
typedef __attribute__((ext_vector_type(8)))  float    v8f;

constexpr int BATCH = 4;
constexpr int LSEQ  = 1024;  // sequence length
constexpr int NH    = 16;    // heads
constexpr int HU    = 64;    // units per head
constexpr int DM    = 1024;  // model dim = NH*HU
constexpr int DIN   = 1024;  // input dim

// k-index mapping for 16-bit A/B fragments of V_WMMA_*_16X16X32_F16 (wave32):
// lane l: row/col = l%16, group g = l/16; element i covers
// k = (i%8) + 8*g + 16*(i/8)   (ISA 7.12.2, 16-bit A-matrix 16x32)
__device__ __forceinline__ int kmap(int i, int g) {
    return (i & 7) + (g << 3) + ((i >> 3) << 4);
}

// Build a v16h fragment from two contiguous 8xf16 groups (k = base+8g..+7 and +16..+23)
__device__ __forceinline__ v16h load_frag_h16(const _Float16* p) {
    const h8 lo = *(const h8*)(p);
    const h8 hi = *(const h8*)(p + 16);
    v16h r;
    #pragma unroll
    for (int i = 0; i < 8; ++i) { r[i] = lo[i]; r[i + 8] = hi[i]; }
    return r;
}

// ---------------------------------------------------------------------------
// Kernel 0: Wt[n*K + k] = (f16) W[k*N + n]   (write-coalesced transpose+convert)
// ---------------------------------------------------------------------------
__global__ __launch_bounds__(256)
void transpose_f16_kernel(const float* __restrict__ W, _Float16* __restrict__ Wt,
                          int K, int N)
{
    const int idx = blockIdx.x * 256 + threadIdx.x;   // total K*N
    const int n = idx / K;
    const int k = idx - n * K;
    Wt[idx] = (_Float16)W[(size_t)k * N + n];
}

// ---------------------------------------------------------------------------
// Kernel 1: Y = X(4096x1024,f32) @ Wt^T (Wt is [n][k], f16) + bias, scaled,
// stored f16 split-head [b][h][l][u] (mode 0) or transposed [b][h][u][l] (mode 1).
// One wave per 16x16 output tile.
// ---------------------------------------------------------------------------
__global__ __launch_bounds__(256)
void qkv_proj_kernel(const float* __restrict__ X,
                     const _Float16* __restrict__ Wt,   // [DM][DIN]
                     const float* __restrict__ bias,
                     _Float16* __restrict__ Yh,
                     float scale, int vtranspose)
{
    const int lane = threadIdx.x & 31;
    const int wave = threadIdx.x >> 5;
    const int tileId = blockIdx.x * 8 + wave;   // 256 mtiles * 64 ntiles
    const int mtile = tileId >> 6;
    const int ntile = tileId & 63;
    const int g  = lane >> 4;
    const int tm = lane & 15;

    const int m = mtile * 16 + tm;              // A row for this lane
    const int n = ntile * 16 + tm;              // B col for this lane
    const float*    xrow  = X  + (size_t)m * DIN;
    const _Float16* wtrow = Wt + (size_t)n * DIN;

    v8f acc = {};
    for (int kb = 0; kb < DIN; kb += 32) {
        if (kb + 64 < DIN) __builtin_prefetch(xrow + kb + 64, 0, 0);
        // A: f32, contiguous groups of 8 at kb+8g and kb+16+8g -> 4x b128
        const float* ap = xrow + kb + 8 * g;
        const float4 f0 = *(const float4*)(ap);
        const float4 f1 = *(const float4*)(ap + 4);
        const float4 f2 = *(const float4*)(ap + 16);
        const float4 f3 = *(const float4*)(ap + 20);
        v16h a;
        a[0]  = (_Float16)f0.x; a[1]  = (_Float16)f0.y; a[2]  = (_Float16)f0.z; a[3]  = (_Float16)f0.w;
        a[4]  = (_Float16)f1.x; a[5]  = (_Float16)f1.y; a[6]  = (_Float16)f1.z; a[7]  = (_Float16)f1.w;
        a[8]  = (_Float16)f2.x; a[9]  = (_Float16)f2.y; a[10] = (_Float16)f2.z; a[11] = (_Float16)f2.w;
        a[12] = (_Float16)f3.x; a[13] = (_Float16)f3.y; a[14] = (_Float16)f3.z; a[15] = (_Float16)f3.w;
        // B: f16 transposed, two contiguous 16B groups -> b128
        const v16h b = load_frag_h16(wtrow + kb + 8 * g);
        acc = __builtin_amdgcn_wmma_f32_16x16x32_f16(false, a, false, b,
                                                     (short)0, acc, false, false);
    }

    const float bn = bias[n];
    const int h = n >> 6;
    const int u = n & 63;
    #pragma unroll
    for (int r = 0; r < 8; ++r) {
        const int mrow = mtile * 16 + r + g * 8;
        const int bidx = mrow >> 10;
        const int l    = mrow & 1023;
        const float val = (acc[r] + bn) * scale;
        const size_t o = vtranspose
            ? ((size_t)(bidx * NH + h) * HU + u) * LSEQ + l     // [b][h][u][l]
            : ((size_t)(bidx * NH + h) * LSEQ + l) * HU + u;    // [b][h][l][u]
        Yh[o] = (_Float16)val;
    }
}

// ---------------------------------------------------------------------------
// Kernel 2: one workgroup per (b,h, 16-query strip).
//   S(16x1024) = Qh Kh^T (Q prescaled by 1/8) -> LDS (64KB of the 320KB WGP LDS)
//   exact softmax in LDS, normalized P written once to d_out attn region
//   ctx(16x64) = P @ V via WMMA (V in [u][l] layout -> contiguous B loads),
//   k-split across 8 waves, LDS f32 atomic reduce
// ---------------------------------------------------------------------------
__global__ __launch_bounds__(256)
void attn_kernel(const _Float16* __restrict__ Qh,   // [B*H][L][U] prescaled
                 const _Float16* __restrict__ Kh,   // [B*H][L][U]
                 const _Float16* __restrict__ Vt,   // [B*H][U][L]  (transposed)
                 float* __restrict__ attn_out,      // [B*H][L][L]
                 _Float16* __restrict__ Ctx)        // [B*L][DM] concat layout
{
    __shared__ float S[16 * LSEQ];      // 64 KB logits / probs strip
    __shared__ float red[16][16];
    __shared__ float rowstat[16];
    __shared__ float ctx_s[16 * HU];    // 4 KB ctx accumulator

    const int lane = threadIdx.x & 31;
    const int wave = threadIdx.x >> 5;
    const int g  = lane >> 4;
    const int tm = lane & 15;
    const int bh    = blockIdx.x >> 6;  // 0..63 = b*16+h
    const int qtile = blockIdx.x & 63;

    const _Float16* Qb = Qh + (size_t)bh * LSEQ * HU;
    const _Float16* Kb = Kh + (size_t)bh * LSEQ * HU;
    const _Float16* Vb = Vt + (size_t)bh * HU * LSEQ;

    for (int i = threadIdx.x; i < 16 * HU; i += 256) ctx_s[i] = 0.0f;

    // Q fragments for d=0..31 and d=32..63 (each wave loads its own copy)
    const _Float16* qrow = Qb + (size_t)(qtile * 16 + tm) * HU + 8 * g;
    const v16h aq0 = load_frag_h16(qrow);
    const v16h aq1 = load_frag_h16(qrow + 32);

    // ---- phase 1: logits strip; B column n = key; rows of Kh are contiguous
    for (int kt = wave; kt < 64; kt += 8) {
        const _Float16* krow = Kb + (size_t)(kt * 16 + tm) * HU + 8 * g;
        const v16h b0 = load_frag_h16(krow);
        const v16h b1 = load_frag_h16(krow + 32);
        v8f acc = {};
        acc = __builtin_amdgcn_wmma_f32_16x16x32_f16(false, aq0, false, b0,
                                                     (short)0, acc, false, false);
        acc = __builtin_amdgcn_wmma_f32_16x16x32_f16(false, aq1, false, b1,
                                                     (short)0, acc, false, false);
        #pragma unroll
        for (int r = 0; r < 8; ++r)
            S[(r + g * 8) * LSEQ + kt * 16 + tm] = acc[r];
    }
    __syncthreads();

    // ---- phase 2: exact softmax over each of the 16 rows (1024 wide)
    {
        const int row = threadIdx.x >> 4;   // 0..15
        const int seg = threadIdx.x & 15;   // 16 segments of 64 columns
        float* srow = S + row * LSEQ + seg * 64;

        float mx = -3.4e38f;
        for (int i = 0; i < 64; ++i) mx = fmaxf(mx, srow[i]);
        red[row][seg] = mx;
        __syncthreads();
        if (threadIdx.x < 16) {
            float mm = red[threadIdx.x][0];
            for (int i = 1; i < 16; ++i) mm = fmaxf(mm, red[threadIdx.x][i]);
            rowstat[threadIdx.x] = mm;
        }
        __syncthreads();
        const float rmax = rowstat[row];

        float s = 0.0f;
        for (int i = 0; i < 64; ++i) {
            const float e = __expf(srow[i] - rmax);
            srow[i] = e;
            s += e;
        }
        __syncthreads();
        red[row][seg] = s;
        __syncthreads();
        if (threadIdx.x < 16) {
            float ss = 0.0f;
            for (int i = 0; i < 16; ++i) ss += red[threadIdx.x][i];
            rowstat[threadIdx.x] = 1.0f / ss;
        }
        __syncthreads();
        const float rinv = rowstat[row];

        float* orow = attn_out + ((size_t)bh * LSEQ + qtile * 16 + row) * LSEQ + seg * 64;
        for (int i = 0; i < 64; ++i) {
            const float p = srow[i] * rinv;
            srow[i] = p;        // keep normalized P in LDS for phase 3
            orow[i] = p;        // stream to HBM (the dominant memory cost)
        }
    }
    __syncthreads();

    // ---- phase 3: ctx = P(16x1024) @ V(1024x64); wave w covers k in [128w,128w+128)
    {
        v8f acc[4] = {v8f{}, v8f{}, v8f{}, v8f{}};
        for (int ks = 0; ks < 4; ++ks) {
            const int kbase = wave * 128 + ks * 32;
            // A from LDS: f32 contiguous groups of 8 -> ds_load b128 pairs
            const float* sp = S + tm * LSEQ + kbase + 8 * g;
            const float4 f0 = *(const float4*)(sp);
            const float4 f1 = *(const float4*)(sp + 4);
            const float4 f2 = *(const float4*)(sp + 16);
            const float4 f3 = *(const float4*)(sp + 20);
            v16h a;
            a[0]  = (_Float16)f0.x; a[1]  = (_Float16)f0.y; a[2]  = (_Float16)f0.z; a[3]  = (_Float16)f0.w;
            a[4]  = (_Float16)f1.x; a[5]  = (_Float16)f1.y; a[6]  = (_Float16)f1.z; a[7]  = (_Float16)f1.w;
            a[8]  = (_Float16)f2.x; a[9]  = (_Float16)f2.y; a[10] = (_Float16)f2.z; a[11] = (_Float16)f2.w;
            a[12] = (_Float16)f3.x; a[13] = (_Float16)f3.y; a[14] = (_Float16)f3.z; a[15] = (_Float16)f3.w;
            #pragma unroll
            for (int nt = 0; nt < 4; ++nt) {
                // B column n = output unit u; V row u contiguous along k
                const v16h b = load_frag_h16(Vb + (size_t)(nt * 16 + tm) * LSEQ + kbase + 8 * g);
                acc[nt] = __builtin_amdgcn_wmma_f32_16x16x32_f16(false, a, false, b,
                                                                 (short)0, acc[nt], false, false);
            }
        }
        #pragma unroll
        for (int nt = 0; nt < 4; ++nt)
            #pragma unroll
            for (int r = 0; r < 8; ++r)
                atomicAdd(&ctx_s[(r + g * 8) * HU + nt * 16 + tm], acc[nt][r]);
    }
    __syncthreads();

    // write ctx strip to workspace in (B*L, DM) concat-head layout
    {
        const int b = bh >> 4, h = bh & 15;
        for (int i = threadIdx.x; i < 16 * HU; i += 256) {
            const int row = i >> 6;
            const int u   = i & 63;
            const int l   = qtile * 16 + row;
            Ctx[((size_t)(b * LSEQ + l)) * DM + h * HU + u] = (_Float16)ctx_s[i];
        }
    }
}

// ---------------------------------------------------------------------------
// Kernel 3: out = Ctx(4096x1024,f16) @ wo + bo -> f32, wo pre-transposed f16 [n][k]
// ---------------------------------------------------------------------------
__global__ __launch_bounds__(256)
void out_proj_kernel(const _Float16* __restrict__ Ctx,
                     const _Float16* __restrict__ Wot,   // [HU][DM]
                     const float* __restrict__ bo,
                     float* __restrict__ out)
{
    const int lane = threadIdx.x & 31;
    const int wave = threadIdx.x >> 5;
    const int tileId = blockIdx.x * 8 + wave;  // 256 mtiles * 4 ntiles
    const int mtile = tileId >> 2;
    const int ntile = tileId & 3;
    const int g  = lane >> 4;
    const int tm = lane & 15;
    const int n = ntile * 16 + tm;
    const _Float16* arow = Ctx + (size_t)(mtile * 16 + tm) * DM + 8 * g;
    const _Float16* brow = Wot + (size_t)n * DM + 8 * g;

    v8f acc = {};
    for (int kb = 0; kb < DM; kb += 32) {
        const v16h a = load_frag_h16(arow + kb);
        const v16h b = load_frag_h16(brow + kb);
        acc = __builtin_amdgcn_wmma_f32_16x16x32_f16(false, a, false, b,
                                                     (short)0, acc, false, false);
    }
    const float bn = bo[n];
    #pragma unroll
    for (int r = 0; r < 8; ++r) {
        const int mrow = mtile * 16 + r + g * 8;
        out[(size_t)mrow * HU + n] = acc[r] + bn;
    }
}

// ---------------------------------------------------------------------------
extern "C" void kernel_launch(void* const* d_in, const int* in_sizes, int n_in,
                              void* d_out, int out_size, void* d_ws, size_t ws_size,
                              hipStream_t stream)
{
    const float* v  = (const float*)d_in[0];
    const float* k  = (const float*)d_in[1];
    const float* q  = (const float*)d_in[2];
    const float* wq = (const float*)d_in[3];
    const float* bq = (const float*)d_in[4];
    const float* wk = (const float*)d_in[5];
    const float* bk = (const float*)d_in[6];
    const float* wv = (const float*)d_in[7];
    const float* bv = (const float*)d_in[8];
    const float* wo = (const float*)d_in[9];
    const float* bo = (const float*)d_in[10];

    float* out  = (float*)d_out;                        // (B, L, U)
    float* attn = out + (size_t)BATCH * LSEQ * HU;      // (B, H, L, L)

    const size_t headElems = (size_t)BATCH * NH * LSEQ * HU; // 4M elems
    _Float16* qh  = (_Float16*)d_ws;
    _Float16* kh  = qh + headElems;
    _Float16* vt  = kh + headElems;          // V stored [b][h][u][l]
    _Float16* ctx = vt + headElems;
    _Float16* wtq = ctx + headElems;         // f16 transposed weights
    _Float16* wtk = wtq + (size_t)DM * DIN;
    _Float16* wtv = wtk + (size_t)DM * DIN;
    _Float16* wot = wtv + (size_t)DM * DIN;  // [HU][DM]
    // total ws: 32MB heads/ctx + 6.1MB weights

    const float scale = 0.125f;  // 1/sqrt(U), folded into Q projection

    transpose_f16_kernel<<<(DIN * DM) / 256, 256, 0, stream>>>(wq, wtq, DIN, DM);
    transpose_f16_kernel<<<(DIN * DM) / 256, 256, 0, stream>>>(wk, wtk, DIN, DM);
    transpose_f16_kernel<<<(DIN * DM) / 256, 256, 0, stream>>>(wv, wtv, DIN, DM);
    transpose_f16_kernel<<<(DM * HU) / 256, 256, 0, stream>>>(wo, wot, DM, HU);

    qkv_proj_kernel<<<2048, 256, 0, stream>>>(q, wtq, bq, qh, scale, 0);
    qkv_proj_kernel<<<2048, 256, 0, stream>>>(k, wtk, bk, kh, 1.0f, 0);
    qkv_proj_kernel<<<2048, 256, 0, stream>>>(v, wtv, bv, vt, 1.0f, 1);
    attn_kernel<<<BATCH * NH * (LSEQ / 16), 256, 0, stream>>>(qh, kh, vt, attn, ctx);
    out_proj_kernel<<<128, 256, 0, stream>>>(ctx, wot, bo, out);
}
